// LinearSelfAttention_45380624449614
// MI455X (gfx1250) — matrix operand order repaired and checked by reading, not compiled, not run on hardware
//
#include <hip/hip_runtime.h>

// ---------------------------------------------------------------------------
// Linear self-attention w/ RoPE + ELU feature map, CDNA5 (gfx1250) wave32.
// GEMMs: v_wmma_f32_16x16x32_f16 with double-buffered LDS tiles filled by the
// Tensor Data Mover (tensor_load_to_lds, TENSORcnt) - one DMA op per 8KB tile.
// ---------------------------------------------------------------------------

typedef __attribute__((ext_vector_type(16))) _Float16 v16h;
typedef __attribute__((ext_vector_type(8)))  _Float16 v8h;
typedef __attribute__((ext_vector_type(4)))  _Float16 v4h;
typedef __attribute__((ext_vector_type(8)))  float    v8f;
typedef __attribute__((ext_vector_type(4)))  unsigned u32x4;
typedef __attribute__((ext_vector_type(8)))  int      i32x8;
typedef __attribute__((ext_vector_type(4)))  int      i32x4;

#define B_  4
#define T_  4096
#define D_  1024
#define H_  16
#define HD_ 64

// ---------------------------------------------------------------------------
// fp32 -> fp16 conversion (vectorized x4)
// ---------------------------------------------------------------------------
__global__ void cvt_f32_f16(const float* __restrict__ in,
                            _Float16* __restrict__ out, int n4) {
    int i = blockIdx.x * blockDim.x + threadIdx.x;
    if (i < n4) {
        float4 v = ((const float4*)in)[i];
        v4h h;
        h[0] = (_Float16)v.x; h[1] = (_Float16)v.y;
        h[2] = (_Float16)v.z; h[3] = (_Float16)v.w;
        ((v4h*)out)[i] = h;
    }
}

// ---------------------------------------------------------------------------
// LDS helpers
// ---------------------------------------------------------------------------
__device__ __forceinline__ unsigned lds_off_u32(void* p) {
    // generic -> LDS(addrspace 3) -> 32-bit workgroup-relative LDS byte addr
    return (unsigned)(unsigned long long)(__attribute__((address_space(3))) char*)p;
}

// ---------------------------------------------------------------------------
// Tensor Data Mover: DMA one 64-row x 128-byte tile (row stride ldg*2 bytes)
// from global memory into a contiguous 8KB LDS tile.  data_size = 8B units:
//   tile_dim0 = 16 units/row, tile_dim1 = 64 rows,
//   tensor_dim0_stride = ldg/4 units.  Tracked on TENSORcnt.
// All descriptor fields are wave-uniform -> D# groups live in SGPRs.
// ---------------------------------------------------------------------------
__device__ __forceinline__ void tdm_tile64(const _Float16* __restrict__ g, int ldg,
                                           int row0, int col0, void* lds) {
    unsigned long long ga = (unsigned long long)(g + (size_t)row0 * ldg + col0);
    unsigned stride_u = (unsigned)(ldg >> 2);       // row stride in 8B units
    unsigned td0 = stride_u;                        // tensor dim0 (>= tile_dim0)
    unsigned td1 = 64;                              // tensor dim1 = tile rows

    u32x4 g0;
    g0[0] = 1u;                                     // count=1 (valid), user mode
    g0[1] = lds_off_u32(lds);                       // lds_addr (bytes)
    g0[2] = (unsigned)(ga & 0xffffffffu);           // global_addr[31:0]
    g0[3] = (unsigned)((ga >> 32) & 0x01ffffffu)    // global_addr[56:32]
          | (2u << 30);                             // type = 2 ("image")

    i32x8 g1;
    g1[0] = (int)(3u << 16);                        // wg_mask=0, data_size=3 (8B)
    g1[1] = (int)((td0 & 0xffffu) << 16);           // barrier_addr=0 | td0[15:0]
    g1[2] = (int)(((td0 >> 16) & 0xffffu) | ((td1 & 0xffffu) << 16));
    g1[3] = (int)(((td1 >> 16) & 0xffffu) | (16u << 16));   // td1[31:16] | tile_dim0=16
    g1[4] = (int)64;                                // tile_dim1=64, tile_dim2=0 (2D)
    g1[5] = (int)stride_u;                          // tensor_dim0_stride[31:0]
    g1[6] = 0;                                      // stride[47:32]=0, td1_stride=0
    g1[7] = 0;

    i32x4 z4 = {0, 0, 0, 0};                        // groups 2/3 unused (2D tile)
    i32x8 z8 = {0, 0, 0, 0, 0, 0, 0, 0};
    __builtin_amdgcn_tensor_load_to_lds(g0, g1, z4, z4, z8, 0);
}

// ---------------------------------------------------------------------------
// WMMA fragment loads from a 64x64-half LDS tile (row-major, ld = 64).
// A-frag (16x32 f16): lane L -> row = base + (L&15);
//   elems 0..7  = K[kOff + (L>>4)*8 ..], elems 8..15 = K[kOff+16+(L>>4)*8 ..]
// B-frag (32x16 f16): lane L -> col n = base + (L&15);
//   elems 0..7  = K[kOff + (L>>4)*8 ..], elems 8..15 = K[kOff+16+(L>>4)*8 ..]
// ---------------------------------------------------------------------------
__device__ __forceinline__ v16h frag_a(const _Float16* s, int rowBase, int lane, int kOff) {
    int m = rowBase + (lane & 15);
    int k = kOff + ((lane >> 4) << 3);
    v8h a0 = *(const v8h*)(s + m * 64 + k);
    v8h a1 = *(const v8h*)(s + m * 64 + k + 16);
    v16h r;
#pragma unroll
    for (int i = 0; i < 8; ++i) { r[i] = a0[i]; r[i + 8] = a1[i]; }
    return r;
}

__device__ __forceinline__ v16h frag_b(const _Float16* s, int colBase, int lane, int kOff) {
    int n = colBase + (lane & 15);
    int k = kOff + ((lane >> 4) << 3);
    v8h b0 = *(const v8h*)(s + n * 64 + k);
    v8h b1 = *(const v8h*)(s + n * 64 + k + 16);
    v16h r;
#pragma unroll
    for (int i = 0; i < 8; ++i) { r[i] = b0[i]; r[i + 8] = b1[i]; }
    return r;
}

// ---------------------------------------------------------------------------
// Core tiled GEMM: C[64x64] += A[64xK] * B[64xK]^T (both K-contiguous),
// K-step 64, double-buffered TDM-filled LDS tiles, 128 threads = 4 waves;
// wave w owns rows w*16..w*16+15, all 64 cols (8 WMMAs per iteration).
// Wave 0 issues the TDM ops (EXEC-independent) and waits TENSORcnt; the
// single barrier per iteration publishes the tiles and frees back buffers
// (a wave only passes the barrier after s_wait_dscnt for its frag reads).
// ---------------------------------------------------------------------------
__device__ __forceinline__ void gemm_64x64(const _Float16* __restrict__ Ag, int lda,
                                           const _Float16* __restrict__ Bg, int ldb,
                                           int m0, int n0, int K,
                                           _Float16* As0, _Float16* As1,
                                           _Float16* Bs0, _Float16* Bs1,
                                           v8f c[4]) {
    int wave = threadIdx.x >> 5;
    int lane = threadIdx.x & 31;
    _Float16 *Acur = As0, *Anxt = As1, *Bcur = Bs0, *Bnxt = Bs1;

    if (threadIdx.x < 32) {                     // wave 0: DMA first tiles
        tdm_tile64(Ag, lda, m0, 0, Acur);
        tdm_tile64(Bg, ldb, n0, 0, Bcur);
    }

    for (int kb = 0; kb < K; kb += 64) {
        if (threadIdx.x < 32)
            __builtin_amdgcn_s_wait_tensorcnt(0);   // wave 0's DMAs complete
        __syncthreads();                            // tiles visible to all waves
        if (kb + 64 < K && threadIdx.x < 32) {      // DMA next tiles (back bufs)
            tdm_tile64(Ag, lda, m0, kb + 64, Anxt);
            tdm_tile64(Bg, ldb, n0, kb + 64, Bnxt);
        }
#pragma unroll
        for (int ks = 0; ks < 2; ++ks) {
            v16h a = frag_a(Acur, wave * 16, lane, ks * 32);
#pragma unroll
            for (int j = 0; j < 4; ++j) {
                v16h b = frag_b(Bcur, j * 16, lane, ks * 32);
                c[j] = __builtin_amdgcn_wmma_f32_16x16x32_f16(
                    false, a, false, b, (short)0, c[j], false, false);
            }
        }
        _Float16* t0 = Acur; Acur = Anxt; Anxt = t0;
        _Float16* t1 = Bcur; Bcur = Bnxt; Bnxt = t1;
    }
}

// Scatter accumulators to a 64x64 f32 LDS tile.
// C-frag layout: elem r of lane L -> row (r + 8*(L>>4)), col (L&15).
__device__ __forceinline__ void store_c_lds(float* Cs, const v8f c[4]) {
    int wave = threadIdx.x >> 5;
    int lane = threadIdx.x & 31;
    int rbase = wave * 16 + ((lane >> 4) << 3);
    int ncol  = lane & 15;
#pragma unroll
    for (int j = 0; j < 4; ++j)
#pragma unroll
        for (int r = 0; r < 8; ++r)
            Cs[(rbase + r) * 64 + j * 16 + ncol] = c[j][r];
}

#define ZERO_ACC(c)                                  \
    v8f c[4];                                        \
    _Pragma("unroll") for (int j_ = 0; j_ < 4; ++j_) \
    _Pragma("unroll") for (int r_ = 0; r_ < 8; ++r_) c[j_][r_] = 0.f;

// ---------------------------------------------------------------------------
// Kernel: QKV projection + fused RoPE + ELU+1 feature-map epilogue.
// grid (M/64, 3D/64). n-tile (64 wide) is head-aligned -> RoPE pairs local.
//   q -> qf  [B,H,T,64]   (feature-mapped, scaled)
//   k -> kfT [B,H,64,T]   (feature-mapped, transposed)
//   v -> vT  [B,H,64,T]   (transposed)
// ---------------------------------------------------------------------------
__global__ void qkv_gemm(const _Float16* __restrict__ xb,
                         const _Float16* __restrict__ wqkvb,
                         _Float16* __restrict__ qf,
                         _Float16* __restrict__ kfT,
                         _Float16* __restrict__ vT) {
    __shared__ alignas(16) _Float16 As0[64 * 64], As1[64 * 64];
    __shared__ alignas(16) _Float16 Bs0[64 * 64], Bs1[64 * 64];
    __shared__ alignas(16) float    Cs[64 * 64];
    int m0 = blockIdx.x * 64;
    int n0 = blockIdx.y * 64;

    ZERO_ACC(c)
    gemm_64x64(xb, D_, wqkvb, D_, m0, n0, D_, As0, As1, Bs0, Bs1, c);
    store_c_lds(Cs, c);
    __syncthreads();

    int sec = n0 >> 10;          // 0=q 1=k 2=v
    int h   = (n0 & 1023) >> 6;  // head index (tile spans exactly one head)

    for (int it = threadIdx.x; it < 64 * 32; it += 128) {
        int row = it >> 5;       // local row 0..63
        int d   = it & 31;       // rope-pair index 0..31
        int m = m0 + row;
        int b = m >> 12;
        int t = m & (T_ - 1);
        float e = Cs[row * 64 + d];
        float o = Cs[row * 64 + d + 32];
        if (sec == 2) {
            size_t base = ((size_t)b * H_ + h) * HD_;
            vT[(base + d)      * T_ + t] = (_Float16)e;
            vT[(base + d + 32) * T_ + t] = (_Float16)o;
        } else {
            // inv_freq = 10000^(-d/32) = exp(-d * ln(10000)/32)
            float inv = expf((float)d * (-9.210340371976184f / 32.0f));
            float ang = (float)t * inv;
            float sn = sinf(ang), cs = cosf(ang);
            float re = e * cs - o * sn;
            float ro = e * sn + o * cs;
            if (sec == 0) { re *= 0.125f; ro *= 0.125f; }   // hd^-0.5
            re = (re > 0.f) ? re + 1.f : expf(re);          // elu + 1
            ro = (ro > 0.f) ? ro + 1.f : expf(ro);
            if (sec == 0) {
                size_t base = (((size_t)b * H_ + h) * T_ + t) * HD_;
                qf[base + d]      = (_Float16)re;
                qf[base + d + 32] = (_Float16)ro;
            } else {
                size_t base = ((size_t)b * H_ + h) * HD_;
                kfT[(base + d)      * T_ + t] = (_Float16)re;
                kfT[(base + d + 32) * T_ + t] = (_Float16)ro;
            }
        }
    }
}

// ---------------------------------------------------------------------------
// Kernel: per-head kv = K_feat^T V, stored as kvh[bh][vd][d]  (K = T = 4096)
// ---------------------------------------------------------------------------
__global__ void kv_gemm(const _Float16* __restrict__ vT,
                        const _Float16* __restrict__ kfT,
                        _Float16* __restrict__ kvh) {
    __shared__ alignas(16) _Float16 As0[64 * 64], As1[64 * 64];
    __shared__ alignas(16) _Float16 Bs0[64 * 64], Bs1[64 * 64];
    __shared__ alignas(16) float    Cs[64 * 64];
    int bh = blockIdx.x;
    const _Float16* A  = vT  + (size_t)bh * HD_ * T_;
    const _Float16* Bm = kfT + (size_t)bh * HD_ * T_;

    ZERO_ACC(c)
    gemm_64x64(A, T_, Bm, T_, 0, 0, T_, As0, As1, Bs0, Bs1, c);
    store_c_lds(Cs, c);
    __syncthreads();
    for (int it = threadIdx.x; it < 64 * 64; it += 128)
        kvh[(size_t)bh * 4096 + it] = (_Float16)Cs[it];
}

// k_sum[bh*64 + d] = sum_t kfT[bh][d][t]
__global__ void ksum_kernel(const _Float16* __restrict__ kfT,
                            float* __restrict__ ksum) {
    __shared__ float red[256];
    int row = blockIdx.x;                      // 0 .. B*H*64-1
    const _Float16* p = kfT + (size_t)row * T_;
    float s = 0.f;
    for (int i = threadIdx.x; i < T_; i += 256) s += (float)p[i];
    red[threadIdx.x] = s;
    __syncthreads();
    for (int off = 128; off > 0; off >>= 1) {
        if (threadIdx.x < off) red[threadIdx.x] += red[threadIdx.x + off];
        __syncthreads();
    }
    if (threadIdx.x == 0) ksum[row] = red[0];
}

// denom[bh*T + t] = max(qf[bh][t][:] . ksum[bh][:], 1e-6)
__global__ void denom_kernel(const _Float16* __restrict__ qf,
                             const float* __restrict__ ksum,
                             float* __restrict__ denom) {
    int idx = blockIdx.x * blockDim.x + threadIdx.x;   // bh*T + t
    int bh = idx >> 12;
    const _Float16* q = qf + (size_t)idx * HD_;
    const float* ks = ksum + bh * HD_;
    float s = 0.f;
#pragma unroll
    for (int d = 0; d < HD_; ++d) s += (float)q[d] * ks[d];
    denom[idx] = fmaxf(s, 1e-6f);
}

// ---------------------------------------------------------------------------
// Kernel: out = (q_feat @ kv) / denom  -> attn [B,T,D] f16.  K = 64.
// grid (T/64, B*H)
// ---------------------------------------------------------------------------
__global__ void attn_gemm(const _Float16* __restrict__ qf,
                          const _Float16* __restrict__ kvh,
                          const float* __restrict__ denom,
                          _Float16* __restrict__ attn) {
    __shared__ alignas(16) _Float16 As0[64 * 64], As1[64 * 64];
    __shared__ alignas(16) _Float16 Bs0[64 * 64], Bs1[64 * 64];
    __shared__ alignas(16) float    Cs[64 * 64];
    int t0 = blockIdx.x * 64;
    int bh = blockIdx.y;
    const _Float16* A  = qf  + (size_t)bh * T_ * HD_;   // [T,64]
    const _Float16* Bm = kvh + (size_t)bh * 4096;       // [vd,64] (K-contig)

    ZERO_ACC(c)
    gemm_64x64(A, HD_, Bm, HD_, t0, 0, HD_, As0, As1, Bs0, Bs1, c);
    store_c_lds(Cs, c);
    __syncthreads();

    int b = bh >> 4, h = bh & 15;
    for (int it = threadIdx.x; it < 64 * 64; it += 128) {
        int row = it >> 6, vd = it & 63;
        int t = t0 + row;
        float val = Cs[it] / denom[bh * T_ + t];
        attn[(((size_t)b * T_ + t) * D_) + h * HD_ + vd] = (_Float16)val;
    }
}

// ---------------------------------------------------------------------------
// Kernel: final projection, f32 output written directly from fragments.
// grid (M/64, D/64)
// ---------------------------------------------------------------------------
__global__ void out_gemm(const _Float16* __restrict__ attn,
                         const _Float16* __restrict__ woutb,
                         float* __restrict__ out) {
    __shared__ alignas(16) _Float16 As0[64 * 64], As1[64 * 64];
    __shared__ alignas(16) _Float16 Bs0[64 * 64], Bs1[64 * 64];
    int m0 = blockIdx.x * 64;
    int n0 = blockIdx.y * 64;

    ZERO_ACC(c)
    gemm_64x64(attn, D_, woutb, D_, m0, n0, D_, As0, As1, Bs0, Bs1, c);

    int wave = threadIdx.x >> 5;
    int lane = threadIdx.x & 31;
    int rbase = m0 + wave * 16 + ((lane >> 4) << 3);
#pragma unroll
    for (int j = 0; j < 4; ++j) {
        int col = n0 + j * 16 + (lane & 15);
#pragma unroll
        for (int r = 0; r < 8; ++r)
            out[(size_t)(rbase + r) * D_ + col] = c[j][r];
    }
}

// ---------------------------------------------------------------------------
// Host launch
// ---------------------------------------------------------------------------
extern "C" void kernel_launch(void* const* d_in, const int* in_sizes, int n_in,
                              void* d_out, int out_size, void* d_ws, size_t ws_size,
                              hipStream_t stream) {
    const float* x    = (const float*)d_in[0];   // [4,4096,1024]
    const float* wqkv = (const float*)d_in[1];   // [3072,1024]
    const float* wout = (const float*)d_in[2];   // [1024,1024]
    float* out = (float*)d_out;                  // [4,4096,1024] f32

    char* ws = (char*)d_ws;
    const size_t M = (size_t)B_ * T_;            // 16384
    _Float16* xb    = (_Float16*)(ws);                         // 32 MB
    _Float16* wqkvb = (_Float16*)(ws + 33554432);              //  6 MB
    _Float16* woutb = (_Float16*)(ws + 39845888);              //  2 MB
    _Float16* qf    = (_Float16*)(ws + 41943040);              // 32 MB
    _Float16* kfT   = (_Float16*)(ws + 75497472);              // 32 MB
    _Float16* vT    = (_Float16*)(ws + 109051904);             // 32 MB
    _Float16* kvh   = (_Float16*)(ws + 142606336);             // 0.5 MB
    float*    ksum  = (float*)(ws + 143130624);                // 16 KB
    float*    denom = (float*)(ws + 143147008);                //  1 MB
    _Float16* attn  = (_Float16*)(ws + 144195584);             // 32 MB

    // 1) f32 -> f16 converts
    cvt_f32_f16<<<(int)(M * D_ / 4 / 256), 256, 0, stream>>>(x, xb, (int)(M * D_ / 4));
    cvt_f32_f16<<<3 * D_ * D_ / 4 / 256, 256, 0, stream>>>(wqkv, wqkvb, 3 * D_ * D_ / 4);
    cvt_f32_f16<<<D_ * D_ / 4 / 256, 256, 0, stream>>>(wout, woutb, D_ * D_ / 4);

    // 2) QKV GEMM + RoPE + ELU feature map (fused epilogue)
    qkv_gemm<<<dim3((int)(M / 64), 3 * D_ / 64), 128, 0, stream>>>(xb, wqkvb, qf, kfT, vT);

    // 3) k_sum, kv GEMM, denom
    ksum_kernel<<<B_ * H_ * HD_, 256, 0, stream>>>(kfT, ksum);
    kv_gemm<<<B_ * H_, 128, 0, stream>>>(vT, kfT, kvh);
    denom_kernel<<<B_ * H_ * T_ / 256, 256, 0, stream>>>(qf, ksum, denom);

    // 4) attention output (fused 1/denom)
    attn_gemm<<<dim3(T_ / 64, B_ * H_), 128, 0, stream>>>(qf, kvh, denom, attn);

    // 5) final projection -> f32 out
    out_gemm<<<dim3((int)(M / 64), D_ / 64), 128, 0, stream>>>(attn, woutb, out);
}
